// GATClassifier_5772436046308
// MI455X (gfx1250) — matrix operand order repaired and checked by reading, not compiled, not run on hardware
//
#include <hip/hip_runtime.h>

typedef __attribute__((ext_vector_type(16))) __bf16 v16bf;
typedef __attribute__((ext_vector_type(8)))  float  v8f;

static inline int cdiv(long long a, long long b){ return (int)((a + b - 1) / b); }

__device__ __forceinline__ unsigned f32_ord(float f){
  unsigned u = __float_as_uint(f);
  return (u & 0x80000000u) ? ~u : (u | 0x80000000u);
}
__device__ __forceinline__ float f32_unord(unsigned e){
  unsigned u = (e & 0x80000000u) ? (e & 0x7fffffffu) : ~e;
  return __uint_as_float(u);
}

// ---- WMMA fragment loaders (f32 memory -> bf16 lane fragments) --------------
// A fragment (16-bit A 16x32, contiguous K order used here): lane L -> row L%16, K-half kh=L/16;
//   elem j in 0..15 -> K = 16*kh + j
// p must point at Arow + k0 + 16*kh.
__device__ __forceinline__ v16bf load_afrag(const float* __restrict__ p){
  v16bf f;
#pragma unroll
  for (int j = 0; j < 8; ++j) { f[j] = (__bf16)p[j]; f[j + 8] = (__bf16)p[8 + j]; }
  return f;
}
// B fragment (32x16): lane L -> col L%16; elem j -> K = 16*kh + j.
// p must point at Wrow + k0 + 16*kh.
__device__ __forceinline__ v16bf load_bfrag(const float* __restrict__ p){
  v16bf f;
#pragma unroll
  for (int j = 0; j < 16; ++j) f[j] = (__bf16)p[j];
  return f;
}

// -----------------------------------------------------------------------------
// C[M x Nn] = A[M x K] @ W[Nn x K]^T + bias   (f32 in/out, bf16 WMMA inside)
// One wave per 32x32 output block: 2x2 register-blocked 16x16 tiles,
// 4 x v_wmma_f32_16x16x32_bf16 per 32-wide K step (independent accumulators).
// Requires M % 32 == 0, Nn % 32 == 0, K % 32 == 0 (true for all calls here).
// -----------------------------------------------------------------------------
__global__ void gemm_bias_wmma(const float* __restrict__ A,
                               const float* __restrict__ W,
                               const float* __restrict__ bias,
                               float* __restrict__ C,
                               int M, int K, int Nn)
{
  int gwave   = (int)((blockIdx.x * blockDim.x + threadIdx.x) >> 5);
  int lane    = threadIdx.x & 31;
  int tilesN2 = Nn >> 5;
  int tiles   = (M >> 5) * tilesN2;
  if (gwave >= tiles) return;
  int tm = (gwave / tilesN2) * 2;     // base 16-row tile index
  int tn = (gwave - (gwave / tilesN2) * tilesN2) * 2;
  int r  = lane & 15;
  int kh = lane >> 4;

  const float* Arow0 = A + (size_t)(tm * 16 + r) * K;
  const float* Arow1 = Arow0 + (size_t)16 * K;
  const float* Wrow0 = W + (size_t)(tn * 16 + r) * K;
  const float* Wrow1 = Wrow0 + (size_t)16 * K;

  v8f acc00 = {}, acc01 = {}, acc10 = {}, acc11 = {};
  for (int k0 = 0; k0 < K; k0 += 32) {
    if (k0 + 32 < K) {               // prefetch next K slice of A
      __builtin_prefetch(Arow0 + k0 + 32, 0, 1);
      __builtin_prefetch(Arow1 + k0 + 32, 0, 1);
    }
    v16bf a0 = load_afrag(Arow0 + k0 + 16 * kh);
    v16bf a1 = load_afrag(Arow1 + k0 + 16 * kh);
    v16bf b0 = load_bfrag(Wrow0 + k0 + 16 * kh);
    v16bf b1 = load_bfrag(Wrow1 + k0 + 16 * kh);
    acc00 = __builtin_amdgcn_wmma_f32_16x16x32_bf16(false, a0, false, b0, (short)0, acc00, false, false);
    acc01 = __builtin_amdgcn_wmma_f32_16x16x32_bf16(false, a0, false, b1, (short)0, acc01, false, false);
    acc10 = __builtin_amdgcn_wmma_f32_16x16x32_bf16(false, a1, false, b0, (short)0, acc10, false, false);
    acc11 = __builtin_amdgcn_wmma_f32_16x16x32_bf16(false, a1, false, b1, (short)0, acc11, false, false);
  }

  // C/D f32 16x16 layout: elem v -> row = v + 8*kh, col = lane%16
  int c0 = tn * 16 + r, c1 = c0 + 16;
  float bb0 = bias ? bias[c0] : 0.0f;
  float bb1 = bias ? bias[c1] : 0.0f;
#pragma unroll
  for (int v = 0; v < 8; ++v) {
    size_t row0 = (size_t)(tm * 16 + v + 8 * kh) * Nn;
    size_t row1 = row0 + (size_t)16 * Nn;
    C[row0 + c0] = acc00[v] + bb0;
    C[row0 + c1] = acc01[v] + bb1;
    C[row1 + c0] = acc10[v] + bb0;
    C[row1 + c1] = acc11[v] + bb1;
  }
}

// -----------------------------------------------------------------------------
// Attention state init: segment-max (ordered-uint -inf), denom, agg output
// -----------------------------------------------------------------------------
__global__ void init_attn(unsigned* __restrict__ menc, int nM,
                          float* __restrict__ den, int nD,
                          float* __restrict__ agg, int nA)
{
  int t = blockIdx.x * blockDim.x + threadIdx.x;
  if (t < nM) menc[t] = 0x007fffffu;  // enc(-inf)
  if (t < nD) den[t]  = 0.0f;
  if (t < nA) agg[t]  = 0.0f;
}

__device__ __forceinline__ void edge_sd(const int* ei, int E, int e, int& s, int& d){
  if (e < E) { s = ei[e]; d = ei[E + e]; } else { s = e - E; d = e - E; }
}

// Pass 1: logit[e,h] = sum_c att[h,c]*leakyrelu(xl[s,h,c]+xr[d,h,c]); segment max
__global__ void edge_logits_kernel(const float* __restrict__ xl,
                                   const float* __restrict__ xr,
                                   const float* __restrict__ att,
                                   const int* __restrict__ ei,
                                   int E, int Etot, int H, int C,
                                   float* __restrict__ logit,
                                   unsigned* __restrict__ menc)
{
  int t = blockIdx.x * blockDim.x + threadIdx.x;
  if (t >= Etot * H) return;
  int e = t / H, h = t - e * H;
  int s, d; edge_sd(ei, E, e, s, d);
  const float4* pl = (const float4*)(xl + (size_t)s * H * C + (size_t)h * C);
  const float4* pr = (const float4*)(xr + (size_t)d * H * C + (size_t)h * C);
  const float4* pa = (const float4*)(att + (size_t)h * C);
  float acc = 0.0f;
  int C4 = C >> 2;
  for (int q = 0; q < C4; ++q) {
    float4 l = pl[q], rr = pr[q], a = pa[q];
    float v0 = l.x + rr.x, v1 = l.y + rr.y, v2 = l.z + rr.z, v3 = l.w + rr.w;
    v0 = (v0 > 0.0f) ? v0 : 0.2f * v0;
    v1 = (v1 > 0.0f) ? v1 : 0.2f * v1;
    v2 = (v2 > 0.0f) ? v2 : 0.2f * v2;
    v3 = (v3 > 0.0f) ? v3 : 0.2f * v3;
    acc += a.x * v0 + a.y * v1 + a.z * v2 + a.w * v3;
  }
  logit[t] = acc;
  atomicMax(&menc[(size_t)d * H + h], f32_ord(acc));
}

// Pass 2: p = exp(logit - m[dst]); denom[dst] += p  (p stored over logit)
__global__ void edge_softmax_kernel(float* __restrict__ logit,
                                    const unsigned* __restrict__ menc,
                                    float* __restrict__ denom,
                                    const int* __restrict__ ei,
                                    int E, int Etot, int H)
{
  int t = blockIdx.x * blockDim.x + threadIdx.x;
  if (t >= Etot * H) return;
  int e = t / H, h = t - e * H;
  int s, d; edge_sd(ei, E, e, s, d);
  float m = f32_unord(menc[(size_t)d * H + h]);
  float p = __expf(logit[t] - m);
  logit[t] = p;
  atomicAdd(&denom[(size_t)d * H + h], p);
}

// Pass 3: agg[dst,h,c] += (p/denom[dst,h]) * xl[src,h,c]; 4 channels per thread
__global__ void edge_agg_kernel(const float* __restrict__ p,
                                const float* __restrict__ denom,
                                const float* __restrict__ xl,
                                const int* __restrict__ ei,
                                int E, int Etot, int H, int C,
                                float* __restrict__ agg)
{
  int HC  = H * C;
  int HC4 = HC >> 2;
  int t = blockIdx.x * blockDim.x + threadIdx.x;
  if (t >= Etot * HC4) return;
  int e = t / HC4, q = t - e * HC4;
  int c0 = q << 2;
  int h  = c0 / C;                       // C % 4 == 0, group never crosses heads
  int s, d; edge_sd(ei, E, e, s, d);
  float a = p[(size_t)e * H + h] / (denom[(size_t)d * H + h] + 1e-16f);
  float4 xv = *(const float4*)(xl + (size_t)s * HC + c0);
  float* o = agg + (size_t)d * HC + c0;
  atomicAdd(o + 0, a * xv.x);
  atomicAdd(o + 1, a * xv.y);
  atomicAdd(o + 2, a * xv.z);
  atomicAdd(o + 3, a * xv.w);
}

// LayerNorm(+bias, +optional residual) then optional ELU, over 256 channels
__global__ void ln_elu_kernel(const float* __restrict__ inp,
                              const float* __restrict__ bias,
                              const float* __restrict__ res,
                              const float* __restrict__ nw,
                              const float* __restrict__ nb,
                              float* __restrict__ dst, int do_elu)
{
  __shared__ float sred[256];
  int n = blockIdx.x, c = threadIdx.x;
  size_t base = (size_t)n * 256;
  float v = inp[base + c] + bias[c];
  if (res) v += res[base + c];
  sred[c] = v; __syncthreads();
  for (int s = 128; s > 0; s >>= 1) { if (c < s) sred[c] += sred[c + s]; __syncthreads(); }
  float mu = sred[0] * (1.0f / 256.0f); __syncthreads();
  float dd = v - mu;
  sred[c] = dd * dd; __syncthreads();
  for (int s = 128; s > 0; s >>= 1) { if (c < s) sred[c] += sred[c + s]; __syncthreads(); }
  float var = sred[0] * (1.0f / 256.0f);
  float hv = dd * rsqrtf(var + 1e-5f) * nw[c] + nb[c];
  if (do_elu) hv = (hv > 0.0f) ? hv : (__expf(hv) - 1.0f);
  dst[base + c] = hv;
}

// Layer-4 tail for the 8 summary nodes only: +b4, LN(32), dot lin_w + lin_b
__global__ void final_head(const float* __restrict__ agg,
                           const float* __restrict__ b4,
                           const float* __restrict__ n4w,
                           const float* __restrict__ n4b,
                           const float* __restrict__ lw,
                           const float* __restrict__ lb,
                           const int* __restrict__ ptr,
                           float* __restrict__ out)
{
  int g = blockIdx.x, c = threadIdx.x;  // 32 threads
  int node = ptr[g];
  float v = agg[(size_t)node * 32 + c] + b4[c];
  float s = v;
  for (int m = 16; m > 0; m >>= 1) s += __shfl_xor(s, m, 32);
  float mu = s * (1.0f / 32.0f);
  float d = v - mu;
  float s2 = d * d;
  for (int m = 16; m > 0; m >>= 1) s2 += __shfl_xor(s2, m, 32);
  float hn = d * rsqrtf(s2 * (1.0f / 32.0f) + 1e-5f) * n4w[c] + n4b[c];
  float p = hn * lw[c];
  for (int m = 16; m > 0; m >>= 1) p += __shfl_xor(p, m, 32);
  if (c == 0) out[g] = p + lb[0];
}

// -----------------------------------------------------------------------------
extern "C" void kernel_launch(void* const* d_in, const int* in_sizes, int n_in,
                              void* d_out, int out_size, void* d_ws, size_t ws_size,
                              hipStream_t stream)
{
  (void)n_in; (void)ws_size;
  const float* x    = (const float*)d_in[0];
  const int*   ei   = (const int*)  d_in[1];
  const int*   ptr  = (const int*)  d_in[2];
  const float* tp_w = (const float*)d_in[3];
  const float* tp_b = (const float*)d_in[4];
  const float* wl1  = (const float*)d_in[5];
  const float* wr1  = (const float*)d_in[6];
  const float* att1 = (const float*)d_in[7];
  const float* b1   = (const float*)d_in[8];
  const float* wl2  = (const float*)d_in[9];
  const float* wr2  = (const float*)d_in[10];
  const float* att2 = (const float*)d_in[11];
  const float* b2   = (const float*)d_in[12];
  const float* wl3  = (const float*)d_in[13];
  const float* wr3  = (const float*)d_in[14];
  const float* att3 = (const float*)d_in[15];
  const float* b3   = (const float*)d_in[16];
  const float* wl4  = (const float*)d_in[17];
  const float* wr4  = (const float*)d_in[18];
  const float* att4 = (const float*)d_in[19];
  const float* b4   = (const float*)d_in[20];
  const float* n1w  = (const float*)d_in[21];
  const float* n1b  = (const float*)d_in[22];
  const float* n2w  = (const float*)d_in[23];
  const float* n2b  = (const float*)d_in[24];
  const float* n3w  = (const float*)d_in[25];
  const float* n3b  = (const float*)d_in[26];
  const float* n4w  = (const float*)d_in[27];
  const float* n4b  = (const float*)d_in[28];
  const float* linw = (const float*)d_in[29];
  const float* linb = (const float*)d_in[30];

  const int TD = 768, HID = 32, HEADS = 8, HC = 256;
  const int N = in_sizes[0] / TD;
  const int E = in_sizes[1] / 2;
  const int Etot = E + N;

  float*    h     = (float*)d_ws;                       // N*HC (layer0 uses N*HID)
  float*    xl    = h    + (size_t)N * HC;              // N*HC
  float*    xr    = xl   + (size_t)N * HC;              // N*HC
  float*    agg   = xr   + (size_t)N * HC;              // N*HC
  float*    logit = agg  + (size_t)N * HC;              // Etot*HEADS
  unsigned* menc  = (unsigned*)(logit + (size_t)Etot * HEADS); // N*HEADS
  float*    denom = (float*)(menc + (size_t)N * HEADS); // N*HEADS

  auto gemm = [&](const float* A, const float* W, const float* bias,
                  float* Cout, int M, int K, int Nn){
    long long waves = (long long)(M / 32) * (Nn / 32);
    gemm_bias_wmma<<<cdiv(waves * 32, 256), 256, 0, stream>>>(A, W, bias, Cout, M, K, Nn);
  };
  auto edges = [&](const float* att, int H, int C){
    long long nm = (long long)N * H, na = (long long)N * H * C;
    init_attn<<<cdiv(na, 256), 256, 0, stream>>>(menc, (int)nm, denom, (int)nm, agg, (int)na);
    long long eh = (long long)Etot * H;
    edge_logits_kernel<<<cdiv(eh, 256), 256, 0, stream>>>(xl, xr, att, ei, E, Etot, H, C, logit, menc);
    edge_softmax_kernel<<<cdiv(eh, 256), 256, 0, stream>>>(logit, menc, denom, ei, E, Etot, H);
    long long ehc4 = (long long)Etot * H * C / 4;
    edge_agg_kernel<<<cdiv(ehc4, 256), 256, 0, stream>>>(logit, denom, xl, ei, E, Etot, H, C, agg);
  };

  // h0 = x @ tp_w^T + tp_b                    (N x 32)
  gemm(x, tp_w, tp_b, h, N, TD, HID);

  // Layer 1 (K=32 -> HC=256), no residual
  gemm(h, wl1, nullptr, xl, N, HID, HC);
  gemm(h, wr1, nullptr, xr, N, HID, HC);
  edges(att1, HEADS, HID);
  ln_elu_kernel<<<N, 256, 0, stream>>>(agg, b1, nullptr, n1w, n1b, h, 1);

  // Layer 2 (K=256 -> 256), residual
  gemm(h, wl2, nullptr, xl, N, HC, HC);
  gemm(h, wr2, nullptr, xr, N, HC, HC);
  edges(att2, HEADS, HID);
  ln_elu_kernel<<<N, 256, 0, stream>>>(agg, b2, h, n2w, n2b, h, 1);

  // Layer 3 (K=256 -> 256), residual
  gemm(h, wl3, nullptr, xl, N, HC, HC);
  gemm(h, wr3, nullptr, xr, N, HC, HC);
  edges(att3, HEADS, HID);
  ln_elu_kernel<<<N, 256, 0, stream>>>(agg, b3, h, n3w, n3b, h, 1);

  // Layer 4 (K=256 -> 32, 1 head), mean over 1 head == identity
  gemm(h, wl4, nullptr, xl, N, HC, HID);
  gemm(h, wr4, nullptr, xr, N, HC, HID);
  edges(att4, 1, HID);

  // Only the `ptr` summary nodes need LN4 + linear head
  final_head<<<out_size, 32, 0, stream>>>(agg, b4, n4w, n4b, linw, linb, ptr, (float*)d_out);
}
